// LightweightAttentionBlock_84739704750238
// MI455X (gfx1250) — matrix operand order repaired and checked by reading, not compiled, not run on hardware
//
#include <hip/hip_runtime.h>
#include <hip/hip_bf16.h>

typedef unsigned short u16;
typedef __attribute__((ext_vector_type(16))) __bf16 v16bf;
typedef __attribute__((ext_vector_type(8)))  float  v8f;
typedef __attribute__((ext_vector_type(4)))  unsigned int v4u;
typedef __attribute__((ext_vector_type(8)))  int v8i;
typedef __attribute__((ext_vector_type(4)))  int v4i;

#define B_DIM 4
#define L_DIM 2048
#define W_DIM 1024
#define F_DIM 4096
#define H_NUM 16
#define D_HEAD 64

union Frag16 { v16bf v; uint4 u[2]; };

__device__ __forceinline__ u16 f2bu(float f) {
  unsigned u = __float_as_uint(f);
  u += 0x7FFFu + ((u >> 16) & 1u);   // round-to-nearest-even
  return (u16)(u >> 16);
}

__device__ __forceinline__ v8f v8zero() {
  v8f z = {0.f,0.f,0.f,0.f,0.f,0.f,0.f,0.f};
  return z;
}

// ------------------------------------------------------------------ TDM
// Issue a 2D tensor DMA: global (row-major, strideElems between rows) ->
// LDS, tile = tileW x tileH elements of 2 bytes, optional LDS row padding
// (ISA 8.3/8.4 D# layout; tensor dims == tile dims so OOB never triggers).
// Wave-uniform; call from ONE wave only; completion via s_wait_tensorcnt.
__device__ __forceinline__ void tdm_load_2d_bf16(
    unsigned ldsByteAddr, const void* gaddr,
    unsigned tileW, unsigned tileH, unsigned strideElems,
    unsigned padEnable, unsigned padIntervalCode, unsigned padAmountCode)
{
  unsigned long long ga = (unsigned long long)gaddr;
  v4u g0;
  g0.x = 1u;                                     // count=1 (valid), user mode
  g0.y = ldsByteAddr;                            // lds_addr (bytes)
  g0.z = (unsigned)(ga & 0xFFFFFFFFu);           // global_addr[31:0]
  g0.w = (unsigned)((ga >> 32) & 0x01FFFFFFu)    // global_addr[56:32]
         | (2u << 30);                           // type = 2 ("image")
  v8i g1;
  g1[0] = (int)((1u << 16)                       // data_size = 1 (2 bytes)
         | (padEnable << 20)                     // pad_enable
         | (padIntervalCode << 22)               // pad_interval (dwords, log)
         | (padAmountCode << 25));               // pad_amount (dwords-1)
  g1[1] = (int)((tileW & 0xFFFFu) << 16);        // tensor_dim0[15:0] @63:48
  g1[2] = (int)(((tileW >> 16) & 0xFFFFu)        // tensor_dim0[31:16]
         | ((tileH & 0xFFFFu) << 16));           // tensor_dim1[15:0]
  g1[3] = (int)(((tileH >> 16) & 0xFFFFu)        // tensor_dim1[31:16]
         | ((tileW & 0xFFFFu) << 16));           // tile_dim0
  g1[4] = (int)(tileH & 0xFFFFu);                // tile_dim1 (tile_dim2 = 0)
  g1[5] = (int)strideElems;                      // tensor_dim0_stride[31:0]
  g1[6] = 0;                                     // stride hi, dim1_stride lo
  g1[7] = 0;
  v4i z4 = {0, 0, 0, 0};
  v8i z8 = {0, 0, 0, 0, 0, 0, 0, 0};
  __builtin_amdgcn_tensor_load_to_lds(g0, g1, z4, z4, z8, 0);
}

// ---------------------------------------------------------------- conversions
__global__ void f32_to_bf16_kernel(const float* __restrict__ in,
                                   u16* __restrict__ out, size_t n4) {
  size_t i = (size_t)blockIdx.x * blockDim.x + threadIdx.x;
  if (i >= n4) return;
  float4 f = ((const float4*)in)[i];
  ushort4 o;
  o.x = f2bu(f.x); o.y = f2bu(f.y); o.z = f2bu(f.z); o.w = f2bu(f.w);
  ((ushort4*)out)[i] = o;
}

// ---------------------------------------------------------------- bf16 GEMM
// C[M,N] = A[M,K] @ B[K,N], row-major bf16 in, fp32 accumulate.
// 128x128 tile, 8 waves, per wave 32x64 = 2x4 WMMA fragments.
// Double-buffered LDS, one barrier/step. A tile arrives by TDM (wave 0,
// pad_enable reproduces the padded LDS stride), B tile is transposed
// through registers.
#define GBM 128
#define GBN 128
#define GBK 32
#define LSTR 40   // padded LDS row stride (elements) = 64B data + 16B pad

template<bool BIAS, bool GELU, bool RES, bool OUTF, bool OUTB>
__global__ __launch_bounds__(256)
void gemm_bf16_kernel(const u16* __restrict__ A, const u16* __restrict__ B,
                      int M, int N, int K,
                      const float* __restrict__ bias,
                      const float* __restrict__ resid,
                      float* __restrict__ outF, u16* __restrict__ outB)
{
  __shared__ u16 sA[2][GBM * LSTR];  // [m][k]   (TDM-filled)
  __shared__ u16 sB[2][GBN * LSTR];  // [n][k]   (transposed B tile)

  const int tid  = threadIdx.x;
  const int wave = tid >> 5;
  const int lane = tid & 31;
  const int ln   = lane & 15;
  const int hi   = lane >> 4;
  const int wm   = wave & 3;      // wave coord along M (x32)
  const int wn   = wave >> 2;     // wave coord along N (x64)
  const int m0   = blockIdx.y * GBM;
  const int n0   = blockIdx.x * GBN;

  int bKk[2], bNb[2];
#pragma unroll
  for (int i = 0; i < 2; ++i) {
    int chunk = tid + i * 256;
    bKk[i] = chunk >> 4;  bNb[i] = (chunk & 15) * 8;
  }

  v8f acc[2][4];
#pragma unroll
  for (int i = 0; i < 2; ++i)
#pragma unroll
    for (int j = 0; j < 4; ++j) acc[i][j] = v8zero();

  const int nk = K / GBK;
  uint4 rb[2];

  // stage tile 0: A via TDM, B via registers
  if (wave == 0)
    tdm_load_2d_bf16((unsigned)(size_t)&sA[0][0], A + (size_t)m0 * K,
                     GBK, GBM, (unsigned)K, 1u, 3u, 3u);  // 64B rows + 16B pad
#pragma unroll
  for (int i = 0; i < 2; ++i)
    rb[i] = *(const uint4*)(B + (size_t)bKk[i] * N + n0 + bNb[i]);
#pragma unroll
  for (int i = 0; i < 2; ++i) {
    const u16* e = (const u16*)&rb[i];
#pragma unroll
    for (int j = 0; j < 8; ++j) sB[0][(bNb[i] + j) * LSTR + bKk[i]] = e[j];
  }

  int cur = 0;
  for (int ks = 0; ks < nk; ++ks) {
    if (wave == 0) __builtin_amdgcn_s_wait_tensorcnt(0);  // A[cur] landed
    __syncthreads();   // publish buf[cur]; buf[cur^1] fully consumed

    const bool hasNext = (ks + 1 < nk);
    if (hasNext) {
      int k1 = (ks + 1) * GBK;
      if (wave == 0)
        tdm_load_2d_bf16((unsigned)(size_t)&sA[cur ^ 1][0],
                         A + (size_t)m0 * K + k1, GBK, GBM, (unsigned)K,
                         1u, 3u, 3u);
#pragma unroll
      for (int i = 0; i < 2; ++i)
        rb[i] = *(const uint4*)(B + (size_t)(k1 + bKk[i]) * N + n0 + bNb[i]);
    }
    if (ks + 2 < nk) {   // L2 hint: global_prefetch_b8
      int k2 = (ks + 2) * GBK;
      __builtin_prefetch(B + (size_t)(k2 + bKk[0]) * N + n0 + bNb[0], 0, 1);
    }

    Frag16 afr[2], bfr[4];
#pragma unroll
    for (int mf = 0; mf < 2; ++mf) {
      int row = wm * 32 + mf * 16 + ln;
      int g0  = hi * 8;                  // A: two groups of 8 consecutive K
      afr[mf].u[0] = *(const uint4*)(&sA[cur][row * LSTR + g0]);
      afr[mf].u[1] = *(const uint4*)(&sA[cur][row * LSTR + g0 + 16]);
    }
#pragma unroll
    for (int nf = 0; nf < 4; ++nf) {
      int n   = wn * 64 + nf * 16 + ln;
      int kk0 = hi * 8;                  // B: two groups of 8 consecutive K
      bfr[nf].u[0] = *(const uint4*)(&sB[cur][n * LSTR + kk0]);
      bfr[nf].u[1] = *(const uint4*)(&sB[cur][n * LSTR + kk0 + 16]);
    }
#pragma unroll
    for (int mf = 0; mf < 2; ++mf)
#pragma unroll
      for (int nf = 0; nf < 4; ++nf)
        acc[mf][nf] = __builtin_amdgcn_wmma_f32_16x16x32_bf16(
            false, afr[mf].v, false, bfr[nf].v, (short)0, acc[mf][nf],
            false, false);

    if (hasNext) {
      int nxt = cur ^ 1;
#pragma unroll
      for (int i = 0; i < 2; ++i) {
        const u16* e = (const u16*)&rb[i];
#pragma unroll
        for (int j = 0; j < 8; ++j) sB[nxt][(bNb[i] + j) * LSTR + bKk[i]] = e[j];
      }
    }
    cur ^= 1;
  }

  // fused epilogue (branch-free via templates)
#pragma unroll
  for (int mf = 0; mf < 2; ++mf)
#pragma unroll
    for (int nf = 0; nf < 4; ++nf)
#pragma unroll
      for (int r = 0; r < 8; ++r) {
        int row = m0 + wm * 32 + mf * 16 + r + hi * 8;
        int col = n0 + wn * 64 + nf * 16 + ln;
        float v = acc[mf][nf][r];
        if constexpr (BIAS) v += bias[col];
        if constexpr (GELU) v = 0.5f * v * (1.0f + erff(v * 0.70710678118654752f));
        size_t off = (size_t)row * N + col;
        if constexpr (RES) v += resid[off];
        if constexpr (OUTF) outF[off] = v;
        if constexpr (OUTB) outB[off] = f2bu(v);
      }
}

// ---------------------------------------------------------------- RoPE split
__global__ __launch_bounds__(256)
void rope_split_kernel(const float* __restrict__ qkv,
                       u16* __restrict__ qb, u16* __restrict__ kb,
                       u16* __restrict__ vb)
{
  size_t idx = (size_t)blockIdx.x * blockDim.x + threadIdx.x;  // < 4<<20
  int j = (int)(idx & 31);
  int l = (int)((idx >> 5) & (L_DIM - 1));
  int h = (int)((idx >> 16) & (H_NUM - 1));
  int b = (int)(idx >> 20);

  size_t base = ((size_t)(b * L_DIM + l)) * (3 * W_DIM) + h * D_HEAD;
  float q1 = qkv[base + j],             q2 = qkv[base + j + 32];
  float k1 = qkv[base + W_DIM + j],     k2 = qkv[base + W_DIM + j + 32];
  float v1 = qkv[base + 2 * W_DIM + j], v2 = qkv[base + 2 * W_DIM + j + 32];

  float invf = __powf(10000.0f, -(float)j * (1.0f / 32.0f));
  float ang  = (float)l * invf;
  float s, c;
  __sincosf(ang, &s, &c);

  size_t ob = ((size_t)((b * H_NUM + h) * L_DIM + l)) * D_HEAD;
  qb[ob + j]      = f2bu(q1 * c - q2 * s);
  qb[ob + j + 32] = f2bu(q2 * c + q1 * s);
  kb[ob + j]      = f2bu(k1 * c - k2 * s);
  kb[ob + j + 32] = f2bu(k2 * c + k1 * s);
  vb[ob + j]      = f2bu(v1);
  vb[ob + j + 32] = f2bu(v2);
}

// ---------------------------------------------------------------- flash attn
// One block = (b, h, 128 q rows). 8 waves x 16 rows. K tile arrives by TDM
// (row-major + padding), V tile transposed through registers. Double-buffered
// LDS, one barrier per 32-wide K/V step.
#define KSTR 72   // sK row stride (elements) = 128B data + 16B pad
#define VSTR 40   // sV row stride
#define PSTR 40   // sP row stride

__global__ __launch_bounds__(256)
void flash_attn_kernel(const u16* __restrict__ qb, const u16* __restrict__ kb,
                       const u16* __restrict__ vb, u16* __restrict__ attnb)
{
  __shared__ u16 sK[2][32 * KSTR];       // [krow][d]   (TDM-filled)
  __shared__ u16 sV[2][D_HEAD * VSTR];   // [d][krow]   (transposed)
  __shared__ u16 sP[8 * 16 * PSTR];      // per-wave P bounce buffer

  const int tid  = threadIdx.x;
  const int wave = tid >> 5, lane = tid & 31;
  const int ln   = lane & 15, hi = lane >> 4;
  const int b = blockIdx.z, h = blockIdx.y;
  const int q0 = blockIdx.x * 128;
  const size_t headbase = ((size_t)(b * H_NUM + h)) * L_DIM * D_HEAD;
  const float scale = 0.125f;  // 1/sqrt(64)

  const int krow = tid >> 3, d8 = (tid & 7) * 8;
  const u16* kHead = kb + headbase;
  const u16* vTile = vb + headbase + (size_t)krow * D_HEAD + d8;

  // Q fragments stay in registers (16 rows x 64 d per wave)
  Frag16 qf[2];
  {
    const u16* qp = qb + headbase + (size_t)(q0 + wave * 16 + ln) * D_HEAD;
    int g0 = hi * 8;
#pragma unroll
    for (int f = 0; f < 2; ++f) {
      qf[f].u[0] = *(const uint4*)(qp + f * 32 + g0);
      qf[f].u[1] = *(const uint4*)(qp + f * 32 + g0 + 16);
    }
  }

  v8f o[4];
#pragma unroll
  for (int f = 0; f < 4; ++f) o[f] = v8zero();
  float mrun[8], lrun[8];
#pragma unroll
  for (int r = 0; r < 8; ++r) { mrun[r] = -3.0e38f; lrun[r] = 0.f; }

  // stage tile 0
  if (wave == 0)
    tdm_load_2d_bf16((unsigned)(size_t)&sK[0][0], kHead,
                     D_HEAD, 32, D_HEAD, 1u, 4u, 3u);  // 128B rows + 16B pad
  uint4 rv = *(const uint4*)(vTile);
  {
    const u16* e = (const u16*)&rv;
#pragma unroll
    for (int j = 0; j < 8; ++j) sV[0][(d8 + j) * VSTR + krow] = e[j];
  }

  int cur = 0;
  for (int kt = 0; kt < L_DIM; kt += 32) {
    if (wave == 0) __builtin_amdgcn_s_wait_tensorcnt(0);  // K[cur] landed
    __syncthreads();   // publish buf[cur]; buf[cur^1] drained

    const bool hasNext = (kt + 32 < L_DIM);
    if (hasNext) {
      if (wave == 0)
        tdm_load_2d_bf16((unsigned)(size_t)&sK[cur ^ 1][0],
                         kHead + (size_t)(kt + 32) * D_HEAD,
                         D_HEAD, 32, D_HEAD, 1u, 4u, 3u);
      rv = *(const uint4*)(vTile + (size_t)(kt + 32) * D_HEAD);
    }
    if (kt + 64 < L_DIM)
      __builtin_prefetch(vTile + (size_t)(kt + 64) * D_HEAD, 0, 1);

    // S = Q @ K^T : 16(q) x 32(k), K-dim = 64
    v8f sf[2];
#pragma unroll
    for (int nf = 0; nf < 2; ++nf) {
      sf[nf] = v8zero();
      int n = nf * 16 + ln;  // k-row = S column
#pragma unroll
      for (int kc = 0; kc < 2; ++kc) {
        Frag16 bf_;
        int doff = kc * 32 + hi * 8;
        bf_.u[0] = *(const uint4*)(&sK[cur][n * KSTR + doff]);
        bf_.u[1] = *(const uint4*)(&sK[cur][n * KSTR + doff + 16]);
        sf[nf] = __builtin_amdgcn_wmma_f32_16x16x32_bf16(
            false, qf[kc].v, false, bf_.v, (short)0, sf[nf], false, false);
      }
    }

    // online softmax; row m = r + 8*hi lives across the 16-lane half group
#pragma unroll
    for (int r = 0; r < 8; ++r) {
      float s0 = sf[0][r] * scale, s1 = sf[1][r] * scale;
      float mx = fmaxf(s0, s1);
#pragma unroll
      for (int off = 8; off >= 1; off >>= 1)
        mx = fmaxf(mx, __shfl_xor(mx, off, 32));
      float mnew = fmaxf(mrun[r], mx);
      float corr = __expf(mrun[r] - mnew);
      float p0 = __expf(s0 - mnew), p1 = __expf(s1 - mnew);
      float rs = p0 + p1;
#pragma unroll
      for (int off = 8; off >= 1; off >>= 1)
        rs += __shfl_xor(rs, off, 32);
      lrun[r] = lrun[r] * corr + rs;
      mrun[r] = mnew;
#pragma unroll
      for (int f = 0; f < 4; ++f) o[f][r] *= corr;
      int prow = r + hi * 8;
      sP[wave * (16 * PSTR) + prow * PSTR + ln]      = f2bu(p0);
      sP[wave * (16 * PSTR) + prow * PSTR + 16 + ln] = f2bu(p1);
    }

    // O += P @ V : repack P through per-wave LDS into A-layout, 4 WMMAs
    Frag16 pf;
    pf.u[0] = *(const uint4*)(&sP[wave * (16 * PSTR) + ln * PSTR + hi * 8]);
    pf.u[1] = *(const uint4*)(&sP[wave * (16 * PSTR) + ln * PSTR + hi * 8 + 16]);
#pragma unroll
    for (int df = 0; df < 4; ++df) {
      Frag16 bf_;
      int n = df * 16 + ln;   // output d column
      bf_.u[0] = *(const uint4*)(&sV[cur][n * VSTR + hi * 8]);
      bf_.u[1] = *(const uint4*)(&sV[cur][n * VSTR + hi * 8 + 16]);
      o[df] = __builtin_amdgcn_wmma_f32_16x16x32_bf16(
          false, pf.v, false, bf_.v, (short)0, o[df], false, false);
    }

    if (hasNext) {
      int nxt = cur ^ 1;
      const u16* e = (const u16*)&rv;
#pragma unroll
      for (int j = 0; j < 8; ++j) sV[nxt][(d8 + j) * VSTR + krow] = e[j];
    }
    cur ^= 1;
  }

  // finalize: O / rowsum, bf16 out in (B, L, H*D) layout for out-proj GEMM
#pragma unroll
  for (int df = 0; df < 4; ++df)
#pragma unroll
    for (int r = 0; r < 8; ++r) {
      int qrow = q0 + wave * 16 + r + hi * 8;
      int col  = h * D_HEAD + df * 16 + ln;
      attnb[((size_t)b * L_DIM + qrow) * W_DIM + col] = f2bu(o[df][r] / lrun[r]);
    }
}

// ---------------------------------------------------------------- launch
extern "C" void kernel_launch(void* const* d_in, const int* in_sizes, int n_in,
                              void* d_out, int out_size, void* d_ws, size_t ws_size,
                              hipStream_t stream) {
  const float* x     = (const float*)d_in[0];
  const float* wqkv  = (const float*)d_in[1];
  const float* wout  = (const float*)d_in[2];
  const float* wffn1 = (const float*)d_in[3];
  const float* bffn1 = (const float*)d_in[4];
  const float* wffn2 = (const float*)d_in[5];
  const float* bffn2 = (const float*)d_in[6];
  float* out = (float*)d_out;

  char* ws = (char*)d_ws;
  size_t off = 0;
  auto alloc = [&](size_t bytes) -> void* {
    void* p = ws + off;
    off += (bytes + 255) & ~(size_t)255;
    return p;
  };
  const size_t BL = (size_t)B_DIM * L_DIM;  // 8192

  u16*   xb     = (u16*)alloc(BL * W_DIM * 2);
  u16*   wqkvb  = (u16*)alloc((size_t)W_DIM * 3 * W_DIM * 2);
  u16*   woutb  = (u16*)alloc((size_t)W_DIM * W_DIM * 2);
  u16*   wffn1b = (u16*)alloc((size_t)W_DIM * F_DIM * 2);
  u16*   wffn2b = (u16*)alloc((size_t)F_DIM * W_DIM * 2);
  float* qkvf   = (float*)alloc(BL * 3 * W_DIM * 4);
  u16*   qb     = (u16*)alloc(BL * W_DIM * 2);
  u16*   kb     = (u16*)alloc(BL * W_DIM * 2);
  u16*   vb     = (u16*)alloc(BL * W_DIM * 2);
  u16*   attnb  = (u16*)alloc(BL * W_DIM * 2);
  float* y1     = (float*)alloc(BL * W_DIM * 4);
  u16*   y1b    = (u16*)alloc(BL * W_DIM * 2);
  u16*   hb     = (u16*)alloc(BL * F_DIM * 2);

  auto cvt = [&](const float* src, u16* dst, size_t n) {
    size_t n4 = n / 4;
    int blocks = (int)((n4 + 255) / 256);
    f32_to_bf16_kernel<<<dim3(blocks), dim3(256), 0, stream>>>(src, dst, n4);
  };
  cvt(x,     xb,     BL * W_DIM);
  cvt(wqkv,  wqkvb,  (size_t)W_DIM * 3 * W_DIM);
  cvt(wout,  woutb,  (size_t)W_DIM * W_DIM);
  cvt(wffn1, wffn1b, (size_t)W_DIM * F_DIM);
  cvt(wffn2, wffn2b, (size_t)F_DIM * W_DIM);

  // qkv = x @ w_qkv  (8192 x 3072, K=1024), fp32 out for RoPE
  gemm_bf16_kernel<false,false,false,true,false>
      <<<dim3(3 * W_DIM / GBN, 8192 / GBM), 256, 0, stream>>>(
      xb, wqkvb, 8192, 3 * W_DIM, W_DIM, nullptr, nullptr, qkvf, nullptr);

  // rotary + split to per-head bf16 q/k/v
  rope_split_kernel<<<dim3((4 << 20) / 256), 256, 0, stream>>>(qkvf, qb, kb, vb);

  // flash attention -> attnb bf16 (B, L, W)
  flash_attn_kernel<<<dim3(L_DIM / 128, H_NUM, B_DIM), 256, 0, stream>>>(
      qb, kb, vb, attnb);

  // y1 = attn @ w_out + x  (fp32 + bf16 copies)
  gemm_bf16_kernel<false,false,true,true,true>
      <<<dim3(W_DIM / GBN, 8192 / GBM), 256, 0, stream>>>(
      attnb, woutb, 8192, W_DIM, W_DIM, nullptr, x, y1, y1b);

  // h = gelu(y1 @ w_ffn1 + b1)  (bf16 only)
  gemm_bf16_kernel<true,true,false,false,true>
      <<<dim3(F_DIM / GBN, 8192 / GBM), 256, 0, stream>>>(
      y1b, wffn1b, 8192, F_DIM, W_DIM, bffn1, nullptr, nullptr, hb);

  // out = h @ w_ffn2 + b2 + y1  (fp32 final)
  gemm_bf16_kernel<true,false,true,true,false>
      <<<dim3(W_DIM / GBN, 8192 / GBM), 256, 0, stream>>>(
      hb, wffn2b, 8192, W_DIM, F_DIM, bffn2, y1, out, nullptr);
}
